// SelfAttention_52097953300545
// MI455X (gfx1250) — matrix operand order repaired and checked by reading, not compiled, not run on hardware
//
#include <hip/hip_runtime.h>

// ---------------------------------------------------------------------------
// Fused QKV-projection + per-head RMSNorm + RoPE for MI455X (gfx1250).
// GEMM: [M=B*T, K=D] x [K=D, N=3D] via v_wmma_f32_16x16x32_bf16 with a
// truncation bf16 hi/lo split (3 WMMAs per K=32 step, ~fp32 accuracy).
// Double-buffered, XOR-swizzled LDS (bank-conflict-minimal b128/b64 traffic),
// one barrier per K-step, global loads + staging overlap WMMA.
// Each 128-col tile of N is one head (Dh=128) -> RMSNorm+RoPE epilogue is
// fused into the GEMM tile, no intermediate buffer.
// ---------------------------------------------------------------------------

typedef __attribute__((ext_vector_type(16))) __bf16 v16bf;
typedef __attribute__((ext_vector_type(8)))  __bf16 v8bf;
typedef __attribute__((ext_vector_type(8)))  float  v8f;
typedef __attribute__((ext_vector_type(4)))  float  v4f;

#define TILE_M 128
#define TILE_N 128
#define TILE_K 32
#define NTHREADS 256
#define PERM_HI16 0x07060302u   // v_perm_b32: {s0[31:16], s1[31:16]}

union Frag { v16bf v; v8bf h[2]; };

// XOR swizzle of the 16B chunk index within a 64B tile row: removes bank
// conflicts while keeping every b128/b64 access naturally aligned.
__device__ __forceinline__ int swz8(int chunk, int row) {
  return (chunk ^ (row >> 2) ^ ((row >> 3) & 2)) & 3;
}

template <int KDIM, int NTOT, int DD, int TT>
__global__ __launch_bounds__(NTHREADS) void fused_qkv_rmsnorm_rope(
    const float* __restrict__ X,   // [M, KDIM]
    const float* __restrict__ W,   // [KDIM, NTOT]
    const float* __restrict__ gq,  // [Dh]
    const float* __restrict__ gk,  // [Dh]
    const float* __restrict__ pos, // [TT]
    float* __restrict__ out,       // q[M,DD] ++ k[M,DD] ++ v[M,DD]
    int Mtot)
{
  // ping-pong tile buffers (swizzled): [buf][row*TILE_K + swzchunk*8 + within]
  __shared__ __bf16 sAhi[2][TILE_M * TILE_K];
  __shared__ __bf16 sAlo[2][TILE_M * TILE_K];
  __shared__ __bf16 sBhi[2][TILE_N * TILE_K];   // transposed: row = n
  __shared__ __bf16 sBlo[2][TILE_N * TILE_K];
  __shared__ float  sRow[4][TILE_M];            // per-column-wave sum(x^2)

  const int tid  = threadIdx.x;
  const int lane = tid & 31;
  const int wave = tid >> 5;
  const int wr   = wave >> 2;   // 0..1 : 64-row band
  const int wc   = wave & 3;    // 0..3 : 16-col slot, paired with slot+64

  const int m0 = blockIdx.y * TILE_M;
  const int n0 = blockIdx.x * TILE_N;

  // ---- staging mappings (all global strides are compile-time constants, so
  //      per-step offsets fold into 24-bit instruction immediates) ----
  const int aRow0 = tid >> 3;          // + 32*i
  const int aCol  = (tid & 7) * 4;     // 4 consecutive K
  const int nB    = tid & 127;         // B: one column n per thread
  const int khalf = tid >> 7;          // 0/1 : lower/upper 4 of each 8-K chunk

  const float* aP = X + (size_t)(m0 + aRow0) * KDIM + aCol;
  const float* bP = W + (size_t)(khalf * 4) * NTOT + (n0 + nB);

  v4f aS[4], bS[4];
  auto loadGlobal = [&]() {
#pragma unroll
    for (int i = 0; i < 4; ++i)
      aS[i] = *(const v4f*)(aP + (size_t)i * 32 * KDIM);
#pragma unroll
    for (int i = 0; i < 4; ++i)
#pragma unroll
      for (int e = 0; e < 4; ++e)
        bS[i][e] = bP[(size_t)(i * 8 + e) * NTOT];   // 4 K-consecutive per col
    aP += TILE_K;
    bP += (size_t)TILE_K * NTOT;
    __builtin_prefetch(aP, 0, 0);   // global_prefetch_b8 (L2-resident data)
    __builtin_prefetch(bP, 0, 0);
  };

  // truncation hi/lo split + v_perm pair packing (~3 VALU per element)
  auto splitPack = [](const v4f& f, uint2& hp, uint2& lp) {
    unsigned uh[4], ul[4];
#pragma unroll
    for (int e = 0; e < 4; ++e) {
      unsigned u = __builtin_bit_cast(unsigned, f[e]);
      float hf = __builtin_bit_cast(float, u & 0xFFFF0000u);
      uh[e] = u;
      ul[e] = __builtin_bit_cast(unsigned, f[e] - hf);
    }
    hp.x = __builtin_amdgcn_perm(uh[1], uh[0], PERM_HI16);
    hp.y = __builtin_amdgcn_perm(uh[3], uh[2], PERM_HI16);
    lp.x = __builtin_amdgcn_perm(ul[1], ul[0], PERM_HI16);
    lp.y = __builtin_amdgcn_perm(ul[3], ul[2], PERM_HI16);
  };

  auto stageToLds = [&](int b) {
#pragma unroll
    for (int i = 0; i < 4; ++i) {
      // A tile: 4 K-consecutive elems -> one aligned b64 hi + one b64 lo
      const int ar = aRow0 + 32 * i;
      uint2 hp, lp;
      splitPack(aS[i], hp, lp);
      const int offA = ar * TILE_K + swz8(aCol >> 3, ar) * 8 + (aCol & 7);
      *(uint2*)(&sAhi[b][offA]) = hp;
      *(uint2*)(&sAlo[b][offA]) = lp;

      // B tile: 4 K-consecutive elems of column nB -> aligned b64 stores
      splitPack(bS[i], hp, lp);
      const int offB = nB * TILE_K + swz8(i, nB) * 8 + khalf * 4;
      *(uint2*)(&sBhi[b][offB]) = hp;
      *(uint2*)(&sBlo[b][offB]) = lp;
    }
  };

  const v8f zero = {0.f, 0.f, 0.f, 0.f, 0.f, 0.f, 0.f, 0.f};
  v8f acc[4][2];
#pragma unroll
  for (int r = 0; r < 4; ++r)
#pragma unroll
    for (int c = 0; c < 2; ++c) acc[r][c] = zero;

  auto computeTile = [&](int b) {
    // Fragment loads per CDNA5 16-bit WMMA VGPR layouts (ISA 7.12.2):
    // A lane L<16: K{0..7,16..23} (chunks h,h+2); lane L>=16: K{8..15,24..31}.
    // B lane n<16: K{0..7,16..23} (chunks 0,2); lane n+16: K{8..15,24..31} (chunks 1,3).
    Frag ahi[4], alo[4], bhi[2], blo[2];
    const int h = lane >> 4;
#pragma unroll
    for (int r = 0; r < 4; ++r) {
      const int row = wr * 64 + r * 16 + (lane & 15);
      ahi[r].h[0] = *(const v8bf*)(&sAhi[b][row * TILE_K + swz8(h,     row) * 8]);
      ahi[r].h[1] = *(const v8bf*)(&sAhi[b][row * TILE_K + swz8(h + 2, row) * 8]);
      alo[r].h[0] = *(const v8bf*)(&sAlo[b][row * TILE_K + swz8(h,     row) * 8]);
      alo[r].h[1] = *(const v8bf*)(&sAlo[b][row * TILE_K + swz8(h + 2, row) * 8]);
    }
#pragma unroll
    for (int c = 0; c < 2; ++c) {
      const int nl = c * 64 + wc * 16 + (lane & 15);
      bhi[c].h[0] = *(const v8bf*)(&sBhi[b][nl * TILE_K + swz8(h,     nl) * 8]);
      bhi[c].h[1] = *(const v8bf*)(&sBhi[b][nl * TILE_K + swz8(h + 2, nl) * 8]);
      blo[c].h[0] = *(const v8bf*)(&sBlo[b][nl * TILE_K + swz8(h,     nl) * 8]);
      blo[c].h[1] = *(const v8bf*)(&sBlo[b][nl * TILE_K + swz8(h + 2, nl) * 8]);
    }
#pragma unroll
    for (int r = 0; r < 4; ++r)
#pragma unroll
      for (int c = 0; c < 2; ++c) {
        acc[r][c] = __builtin_amdgcn_wmma_f32_16x16x32_bf16(
            false, ahi[r].v, false, bhi[c].v, (short)0, acc[r][c], false, false);
        acc[r][c] = __builtin_amdgcn_wmma_f32_16x16x32_bf16(
            false, ahi[r].v, false, blo[c].v, (short)0, acc[r][c], false, false);
        acc[r][c] = __builtin_amdgcn_wmma_f32_16x16x32_bf16(
            false, alo[r].v, false, bhi[c].v, (short)0, acc[r][c], false, false);
      }
  };

  // ---------------- pipelined K-loop (1 barrier per step) ----------------
  const int nk = KDIM / TILE_K;
  loadGlobal();
  stageToLds(0);
  __syncthreads();
  for (int kt = 0; kt < nk; ++kt) {
    const int cur = kt & 1;
    const bool more = (kt + 1) < nk;
    if (more) loadGlobal();        // vmem issues early, overlaps WMMA below
    computeTile(cur);
    if (more) stageToLds(cur ^ 1); // write other buffer while peers compute
    __syncthreads();
  }

  // ---------------- fused epilogue ----------------
  constexpr int tilesPerSec = DD / TILE_N;            // = NUM_HEADS
  const int sec  = blockIdx.x / tilesPerSec;          // 0=q 1=k 2=v
  const int head = blockIdx.x % tilesPerSec;
  const size_t BTD = (size_t)Mtot * (size_t)DD;
  float* base = out + (size_t)sec * BTD + (size_t)head * TILE_N;

  if (sec < 2) {
    const float* g = (sec == 0) ? gq : gk;

    // per-row sum of squares; this wave holds cols {wc*16+l, 64+wc*16+l}
#pragma unroll
    for (int r = 0; r < 4; ++r)
#pragma unroll
      for (int e = 0; e < 8; ++e) {
        float v0 = acc[r][0][e], v1 = acc[r][1][e];
        float s = v0 * v0 + v1 * v1;
        s += __shfl_xor(s, 1);
        s += __shfl_xor(s, 2);
        s += __shfl_xor(s, 4);
        s += __shfl_xor(s, 8);                        // reduce within 16-lane half
        const int row = wr * 64 + r * 16 + e;
        if (lane == 0)       sRow[wc][row]     = s;
        else if (lane == 16) sRow[wc][row + 8] = s;
      }
    __syncthreads();

    const int j  = wc * 16 + (lane & 15);             // rotate pair (j, j+64)
    const float g0 = g[j], g1 = g[j + 64];
    // inv_freq = 10000^(-j/(Dh/2)) ; ln(10000) = 9.2103403719761836
    const float ifreq = __expf((float)j * (-9.210340371976184f / 64.0f));

#pragma unroll
    for (int r = 0; r < 4; ++r)
#pragma unroll
      for (int e = 0; e < 8; ++e) {
        const int row = wr * 64 + r * 16 + ((lane & 16) >> 1) + e;
        const int m = m0 + row;
        const float ss = sRow[0][row] + sRow[1][row] + sRow[2][row] + sRow[3][row];
        const float scale = rsqrtf(ss * (1.0f / (float)TILE_N) + 1e-6f);
        const float p = pos[m % TT];                  // TT is a power of two
        float sn, cs;
        __sincosf(p * ifreq, &sn, &cs);
        const float x1 = acc[r][0][e] * scale * g0;
        const float x2 = acc[r][1][e] * scale * g1;
        float* o = base + (size_t)m * DD;
        o[j]      = x1 * cs - x2 * sn;
        o[j + 64] = x1 * sn + x2 * cs;
      }
  } else {
    // v: passthrough store
#pragma unroll
    for (int r = 0; r < 4; ++r)
#pragma unroll
      for (int c = 0; c < 2; ++c) {
        const int nl = c * 64 + wc * 16 + (lane & 15);
#pragma unroll
        for (int e = 0; e < 8; ++e) {
          const int m = m0 + wr * 64 + r * 16 + ((lane & 16) >> 1) + e;
          base[(size_t)m * DD + nl] = acc[r][c][e];
        }
      }
  }
}

extern "C" void kernel_launch(void* const* d_in, const int* in_sizes, int n_in,
                              void* d_out, int out_size, void* d_ws, size_t ws_size,
                              hipStream_t stream) {
  (void)n_in; (void)out_size; (void)d_ws; (void)ws_size;
  const float* x   = (const float*)d_in[0];
  const float* W   = (const float*)d_in[1];
  const float* gq  = (const float*)d_in[2];
  const float* gk  = (const float*)d_in[3];
  const float* pos = (const float*)d_in[4];
  float* out = (float*)d_out;

  // Harness shape (setup_inputs): B=2, T=4096, D=2048, H=16 -> compile-time
  // strides so global-load offsets fold into instruction immediates.
  constexpr int KDIM = 2048;
  constexpr int NTOT = 3 * 2048;
  constexpr int DD   = 2048;
  constexpr int TT   = 4096;

  const int M = in_sizes[0] / DD;     // B*T = 8192 (runtime)

  dim3 grid(NTOT / TILE_N, M / TILE_M);   // 48 x 64 tiles
  fused_qkv_rmsnorm_rope<KDIM, NTOT, DD, TT><<<grid, NTHREADS, 0, stream>>>(
      x, W, gq, gk, pos, out, M);
}
